// S_GNN_11012296147403
// MI455X (gfx1250) — matrix operand order repaired and checked by reading, not compiled, not run on hardware
//
#include <hip/hip_runtime.h>
#include <math.h>

typedef __attribute__((ext_vector_type(16))) _Float16 v16h;
typedef __attribute__((ext_vector_type(8)))  _Float16 v8h;
typedef __attribute__((ext_vector_type(8)))  float    v8f;

// Problem constants (match reference)
constexpr int BATCH = 512;
constexpr int NN    = 50;
constexpr int FEAT  = 300;
constexpr int DD    = 512;   // D
constexpr int DD2   = 1024;  // 2*D
constexpr int LL    = 2048;
constexpr int MTOK  = BATCH * NN;   // 25600

// d_out layout: s_out (512*50*600) | adj_masked (512*50*50) | att_loss (1)
constexpr size_t OFF_ADJ  = (size_t)BATCH * NN * 600;          // 15,360,000
constexpr size_t OFF_LOSS = OFF_ADJ + (size_t)BATCH * NN * NN; // 16,640,000

// Workspace layout (offsets in _Float16 elements)
constexpr size_t O_WF1  = 0;                       // 512*512
constexpr size_t O_WF2  = O_WF1 + 262144;
constexpr size_t O_WF3  = O_WF2 + 262144;          // 1024*1024
constexpr size_t O_WF4  = O_WF3 + 1048576;
constexpr size_t O_WF5  = O_WF4 + 1048576;
constexpr size_t O_WL1  = O_WF5 + 1048576;         // 1024*2048
constexpr size_t O_WL2  = O_WL1 + 2097152;
constexpr size_t O_WOUT = O_WL2 + 2097152;         // 300*1024
constexpr size_t O_LH   = O_WOUT + 307200;         // 512*2048
constexpr size_t O_LP1  = O_LH + 1048576;          // 512*1024
constexpr size_t O_LP2  = O_LP1 + 524288;
constexpr size_t O_COMB = O_LP2 + 524288;          // 25600*1024
constexpr size_t O_LMS  = O_COMB + 26214400;       // 25600*1024 (later: prepared)
constexpr size_t O_FEA4 = O_LMS + 26214400;        // 25600*1024 (tmp t1, later msg_in)
constexpr size_t H_END  = O_FEA4 + 26214400;       // 88,911,872 halves
constexpr size_t ADJ_BYTE_OFF = H_END * 2;         // f32 adj ws, 512*50*50

// ---------------------------------------------------------------------------
// f32 -> f16 conversion
__global__ void cvt_kernel(const float* __restrict__ src, _Float16* __restrict__ dst, int n) {
    int i = blockIdx.x * blockDim.x + threadIdx.x;
    if (i < n) dst[i] = (_Float16)src[i];
}

__global__ void zero_loss_kernel(float* out) {
    if (threadIdx.x == 0 && blockIdx.x == 0) out[OFF_LOSS] = 0.0f;
}

// ---------------------------------------------------------------------------
// bb features + assemble s_with_bb (f16 into combined[:, 0:512]) + masked s copy to out
__global__ __launch_bounds__(256) void bb_kernel(
    const float* __restrict__ s, const float* __restrict__ ps,
    const int* __restrict__ mask_s,
    const float* __restrict__ W_bb, const float* __restrict__ b_bb,
    _Float16* __restrict__ combined, float* __restrict__ out_s)
{
    const int m = blockIdx.x;           // token index 0..25599
    const int b = m / NN, i = m - b * NN;
    __shared__ float feat[16];
    if (threadIdx.x == 0) {
        float x0 = ps[m * 4 + 0], y0 = ps[m * 4 + 1];
        float x1 = ps[m * 4 + 2], y1 = ps[m * 4 + 3];
        float w = x1 - x0, h = y1 - y0;
        feat[0] = x0; feat[1] = y0; feat[2] = x1; feat[3] = y1;
        feat[4] = w;  feat[5] = h;
        feat[6] = x0 + 0.5f * w; feat[7] = y0 + 0.5f * h;
        feat[8] = w * h; feat[9] = w / (h + 1e-14f);
    }
    __syncthreads();
    const float mk = (i < mask_s[b]) ? 1.0f : 0.0f;
    for (int t = threadIdx.x; t < DD; t += 256) {
        if (t < FEAT) {
            float v = s[(size_t)m * FEAT + t];
            combined[(size_t)m * DD2 + t] = (_Float16)v;
            out_s[(size_t)m * 600 + t] = v * mk;
        } else {
            int n = t - FEAT;   // 0..211
            float acc = b_bb[n];
            #pragma unroll
            for (int k = 0; k < 10; k++) acc += feat[k] * W_bb[n * 10 + k];
            combined[(size_t)m * DD2 + t] = (_Float16)(acc > 0.f ? acc : 0.f);
        }
    }
}

// ---------------------------------------------------------------------------
// Generic WMMA GEMM:  C[M x Ncols] = epi( A[M x K] @ W[Ncols x K]^T + bias )
// Block: 256 thr (8 waves 2Mx4N), block tile 128x128, wave tile 64x32,
// double-buffered LDS (pitch 40 halves, conflict-free fragment reads).
// GUARD: compile-time column bounds check (only needed when Ncols % 128 != 0).
// MODE 0: relu -> f16      MODE 1: relu * g[row*ldg+col] -> f16
// MODE 2: relu * lp[(row/50)*1024+col] -> f16
// MODE 3: relu, mask by (i < mask_s[b]), f32 store to out_s[...,300+col]
template <int MODE, bool GUARD>
__global__ __launch_bounds__(256) void gemm_f16(
    const _Float16* __restrict__ A, int lda,
    const _Float16* __restrict__ Wt, int K,
    const float* __restrict__ bias, int Ncols,
    _Float16* __restrict__ outh, int ldo,
    const _Float16* __restrict__ mul_h, int ldg,
    const int* __restrict__ mask_s, float* __restrict__ outf)
{
    constexpr int PITCH = 40;   // halves per LDS row (32 data + 8 pad)
    __shared__ _Float16 sA[2][128 * PITCH];
    __shared__ _Float16 sW[2][128 * PITCH];

    const int tid  = threadIdx.x;
    const int lane = tid & 31;
    const int wave = tid >> 5;
    const int wm = wave & 1;        // 0..1 (M)
    const int wn = wave >> 1;       // 0..3 (N)
    const int rowBase = blockIdx.y * 128;
    const int colBase = blockIdx.x * 128;

    v8f acc[4][2];
    #pragma unroll
    for (int i = 0; i < 4; i++)
        #pragma unroll
        for (int j = 0; j < 2; j++) acc[i][j] = (v8f){};

    // cooperative tile load: each thread moves 16 halves of A and of W
    const int lr = tid >> 1;              // row 0..127
    const int lc = (tid & 1) << 4;        // 0 or 16 halves
    const _Float16* gA = A + (size_t)(rowBase + lr) * lda + lc;
    const bool wok = !GUARD || (colBase + lr) < Ncols;
    const _Float16* gW = Wt + (size_t)(GUARD ? min(colBase + lr, Ncols - 1)
                                             : (colBase + lr)) * K + lc;

    auto load_tiles = [&](int k0, int buf) {
        {
            const _Float16* gp = gA + k0;
            __builtin_prefetch(gp + 64, 0, 1);
            v8h* d = (v8h*)&sA[buf][lr * PITCH + lc];
            d[0] = *(const v8h*)gp;
            d[1] = *(const v8h*)(gp + 8);
        }
        {
            v8h* d = (v8h*)&sW[buf][lr * PITCH + lc];
            if (wok) {
                const _Float16* gp = gW + k0;
                __builtin_prefetch(gp + 64, 0, 1);
                d[0] = *(const v8h*)gp;
                d[1] = *(const v8h*)(gp + 8);
            } else {
                v8h z = {};
                d[0] = z; d[1] = z;
            }
        }
    };

    load_tiles(0, 0);
    const int nk = K >> 5;
    const int frA_m = lane & 15;
    const int frA_k = (lane < 16) ? 0 : 8;
    const int frB_n = lane & 15;
    const int frB_k = (lane >> 4) << 3;

    for (int ks = 0; ks < nk; ks++) {
        __syncthreads();
        const int cur = ks & 1;
        if (ks + 1 < nk) load_tiles((ks + 1) << 5, cur ^ 1);

        v16h afr[4], bfr[2];
        #pragma unroll
        for (int sm = 0; sm < 4; sm++) {
            int off = (wm * 64 + sm * 16 + frA_m) * PITCH + frA_k;
            v8h lo = *(const v8h*)&sA[cur][off];
            v8h hi = *(const v8h*)&sA[cur][off + 16];
            #pragma unroll
            for (int q = 0; q < 8; q++) { afr[sm][q] = lo[q]; afr[sm][q + 8] = hi[q]; }
        }
        #pragma unroll
        for (int sn = 0; sn < 2; sn++) {
            int off = (wn * 32 + sn * 16 + frB_n) * PITCH + frB_k;
            v8h lo = *(const v8h*)&sW[cur][off];
            v8h hi = *(const v8h*)&sW[cur][off + 16];
            #pragma unroll
            for (int q = 0; q < 8; q++) { bfr[sn][q] = lo[q]; bfr[sn][q + 8] = hi[q]; }
        }
        #pragma unroll
        for (int sm = 0; sm < 4; sm++)
            #pragma unroll
            for (int sn = 0; sn < 2; sn++)
                acc[sm][sn] = __builtin_amdgcn_wmma_f32_16x16x32_f16(
                    false, afr[sm], false, bfr[sn], (short)0, acc[sm][sn], false, false);
    }

    // epilogue
    #pragma unroll
    for (int sm = 0; sm < 4; sm++) {
        const int row0 = rowBase + wm * 64 + sm * 16 + ((lane < 16) ? 0 : 8);
        #pragma unroll
        for (int sn = 0; sn < 2; sn++) {
            const int col = colBase + wn * 32 + sn * 16 + (lane & 15);
            const bool colok = !GUARD || col < Ncols;
            const float bv = colok ? bias[col] : 0.0f;
            #pragma unroll
            for (int r = 0; r < 8; r++) {
                const int row = row0 + r;
                float v = acc[sm][sn][r] + bv;
                v = v > 0.0f ? v : 0.0f;
                if (MODE == 0) {
                    if (colok) outh[(size_t)row * ldo + col] = (_Float16)v;
                } else if (MODE == 1) {
                    if (colok) {
                        v *= (float)mul_h[(size_t)row * ldg + col];
                        outh[(size_t)row * ldo + col] = (_Float16)v;
                    }
                } else if (MODE == 2) {
                    if (colok) {
                        v *= (float)mul_h[(size_t)(row / NN) * DD2 + col];
                        outh[(size_t)row * ldo + col] = (_Float16)v;
                    }
                } else { // MODE 3: masked message store to d_out
                    if (colok) {
                        const int b = row / NN, i = row - b * NN;
                        const float vv = (i < mask_s[b]) ? v : 0.0f;
                        outf[(size_t)b * (NN * 600) + (size_t)i * 600 + FEAT + col] = vv;
                    }
                }
            }
        }
    }
}

// ---------------------------------------------------------------------------
// Per-batch attention: scores = fea4[b] @ lms[b]^T via WMMA (64x64 padded),
// then masked softmax over j, write adj (ws) + adj_masked (out) + att_loss.
// Padded rows/cols (>= 50) are computed from CLAMPED row pointers (garbage but
// finite) and are never read by the softmax phase -> branch-free inner loop.
__global__ __launch_bounds__(512) void adj_kernel(
    const _Float16* __restrict__ fea4, const _Float16* __restrict__ lms,
    const int* __restrict__ mask_s,
    float* __restrict__ adjws, float* __restrict__ out)
{
    const int b = blockIdx.x;
    const int tid = threadIdx.x, lane = tid & 31, wave = tid >> 5;
    const int ti = wave >> 2, tj = wave & 3;   // 4x4 grid of 16x16 tiles

    const _Float16* Af = fea4 + (size_t)b * NN * DD2;
    const _Float16* Bf = lms  + (size_t)b * NN * DD2;
    const int am = min(ti * 16 + (lane & 15), NN - 1);   // clamp: pad region unused
    const int bn = min(tj * 16 + (lane & 15), NN - 1);
    const _Float16* pa = Af + (size_t)am * DD2 + ((lane < 16) ? 0 : 8);
    const _Float16* pb = Bf + (size_t)bn * DD2 + ((lane >> 4) << 3);

    v8f acc = (v8f){};
    for (int k0 = 0; k0 < DD2; k0 += 32) {
        v8h alo = *(const v8h*)(pa + k0);
        v8h ahi = *(const v8h*)(pa + k0 + 16);
        v8h blo = *(const v8h*)(pb + k0);
        v8h bhi = *(const v8h*)(pb + k0 + 16);
        v16h a, w;
        #pragma unroll
        for (int q = 0; q < 8; q++) {
            a[q] = alo[q]; a[q + 8] = ahi[q];
            w[q] = blo[q]; w[q + 8] = bhi[q];
        }
        acc = __builtin_amdgcn_wmma_f32_16x16x32_f16(false, a, false, w, (short)0, acc, false, false);
    }

    __shared__ float sS[64][66];
    __shared__ float snorm[64];
    {
        const int r0 = ti * 16 + ((lane < 16) ? 0 : 8);
        const int c  = tj * 16 + (lane & 15);
        #pragma unroll
        for (int r = 0; r < 8; r++) sS[r0 + r][c] = acc[r];
    }
    __syncthreads();

    if (tid < NN) {
        const int i = tid;
        const int ms = mask_s[b];
        auto mval = [&](int j) -> float {
            bool allowed = (((i > j) ? i : j) < ms) || (i >= ms);
            if (i == j) allowed = false;
            if (ms == 1 && i == 0 && j == 0) allowed = true;
            return allowed ? sS[i][j] : -3.0e38f;
        };
        float mx = -3.0e38f;
        for (int j = 0; j < NN; j++) mx = fmaxf(mx, mval(j));
        float sum = 0.0f;
        for (int j = 0; j < NN; j++) sum += __expf(mval(j) - mx);
        const float inv = 1.0f / sum;
        const float rmk = (i < ms) ? 1.0f : 0.0f;
        float sq = 0.0f;
        for (int j = 0; j < NN; j++) {
            const float aij = __expf(mval(j) - mx) * inv;
            adjws[((size_t)b * NN + i) * NN + j] = aij;
            const float amk = aij * rmk;
            out[OFF_ADJ + ((size_t)b * NN + i) * NN + j] = amk;
            sq += amk * amk;
        }
        snorm[i] = sqrtf(sq);
    }
    __syncthreads();
    if (tid == 0) {
        float fro = 0.0f;
        for (int i = 0; i < NN; i++) fro += snorm[i];
        atomicAdd(&out[OFF_LOSS], -(fro / (float)mask_s[b]) * 0.1f);
    }
}

// ---------------------------------------------------------------------------
// msg_in[b,i,:] = adj[b,i,:] @ prepared[b,:,:]   (K=50, VALU)
__global__ __launch_bounds__(256) void msg_kernel(
    const float* __restrict__ adjws, const _Float16* __restrict__ prepared,
    _Float16* __restrict__ msg_in)
{
    const int m = blockIdx.x;
    const int b = m / NN;
    __shared__ float aj[NN];
    if (threadIdx.x < NN) aj[threadIdx.x] = adjws[(size_t)m * NN + threadIdx.x];
    __syncthreads();
    const _Float16* pb = prepared + (size_t)b * NN * DD2;
    for (int d = threadIdx.x; d < DD2; d += 256) {
        float acc = 0.0f;
        #pragma unroll 5
        for (int j = 0; j < NN; j++) acc += aj[j] * (float)pb[(size_t)j * DD2 + d];
        msg_in[(size_t)m * DD2 + d] = (_Float16)acc;
    }
}

// ---------------------------------------------------------------------------
extern "C" void kernel_launch(void* const* d_in, const int* in_sizes, int n_in,
                              void* d_out, int out_size, void* d_ws, size_t ws_size,
                              hipStream_t stream) {
    const float* l_f32 = (const float*)d_in[0];
    const float* s     = (const float*)d_in[1];
    const float* ps    = (const float*)d_in[2];
    const int*   mask  = (const int*)d_in[3];
    const float* W_bb  = (const float*)d_in[4];
    const float* b_bb  = (const float*)d_in[5];
    const float* W_f1  = (const float*)d_in[6];   const float* b_f1 = (const float*)d_in[7];
    const float* W_f2  = (const float*)d_in[8];   const float* b_f2 = (const float*)d_in[9];
    const float* W_f3  = (const float*)d_in[10];  const float* b_f3 = (const float*)d_in[11];
    const float* W_f4  = (const float*)d_in[12];  const float* b_f4 = (const float*)d_in[13];
    const float* W_f5  = (const float*)d_in[14];  const float* b_f5 = (const float*)d_in[15];
    const float* W_l1  = (const float*)d_in[16];  const float* b_l1 = (const float*)d_in[17];
    const float* W_l2  = (const float*)d_in[18];  const float* b_l2 = (const float*)d_in[19];
    const float* W_out = (const float*)d_in[20];  const float* b_out = (const float*)d_in[21];

    _Float16* ws  = (_Float16*)d_ws;
    float* out    = (float*)d_out;
    float* adjws  = (float*)((char*)d_ws + ADJ_BYTE_OFF);

    auto cvt = [&](const float* src, _Float16* dst, int n) {
        cvt_kernel<<<(n + 255) / 256, 256, 0, stream>>>(src, dst, n);
    };
    cvt(W_f1,  ws + O_WF1,  DD * DD);
    cvt(W_f2,  ws + O_WF2,  DD * DD);
    cvt(W_f3,  ws + O_WF3,  DD2 * DD2);
    cvt(W_f4,  ws + O_WF4,  DD2 * DD2);
    cvt(W_f5,  ws + O_WF5,  DD2 * DD2);
    cvt(W_l1,  ws + O_WL1,  DD2 * LL);
    cvt(W_l2,  ws + O_WL2,  DD2 * LL);
    cvt(W_out, ws + O_WOUT, FEAT * DD2);
    cvt(l_f32, ws + O_LH,   BATCH * LL);

    // s_with_bb -> combined[:, 0:512] (f16); masked s -> out cols 0..299
    bb_kernel<<<MTOK, 256, 0, stream>>>(s, ps, mask, W_bb, b_bb, ws + O_COMB, out);
    zero_loss_kernel<<<1, 32, 0, stream>>>(out);

    // lp1 / lp2: (512 x 2048) @ (1024 x 2048)^T
    gemm_f16<0, false><<<dim3(DD2 / 128, BATCH / 128), 256, 0, stream>>>(
        ws + O_LH, LL, ws + O_WL1, LL, b_l1, DD2, ws + O_LP1, DD2, nullptr, 0, nullptr, nullptr);
    gemm_f16<0, false><<<dim3(DD2 / 128, BATCH / 128), 256, 0, stream>>>(
        ws + O_LH, LL, ws + O_WL2, LL, b_l2, DD2, ws + O_LP2, DD2, nullptr, 0, nullptr, nullptr);

    // t1 = relu(f1(s_with_bb)) -> fea4 buffer (ld 512)
    gemm_f16<0, false><<<dim3(DD / 128, MTOK / 128), 256, 0, stream>>>(
        ws + O_COMB, DD2, ws + O_WF1, DD, b_f1, DD, ws + O_FEA4, DD, nullptr, 0, nullptr, nullptr);
    // gated = relu(f2(..)) * t1 -> combined[:, 512:1024]
    gemm_f16<1, false><<<dim3(DD / 128, MTOK / 128), 256, 0, stream>>>(
        ws + O_COMB, DD2, ws + O_WF2, DD, b_f2, DD, ws + O_COMB + DD, DD2, ws + O_FEA4, DD, nullptr, nullptr);

    // l_masked_source = relu(f3(combined)) * lp1 -> lms
    gemm_f16<2, false><<<dim3(DD2 / 128, MTOK / 128), 256, 0, stream>>>(
        ws + O_COMB, DD2, ws + O_WF3, DD2, b_f3, DD2, ws + O_LMS, DD2, ws + O_LP1, 0, nullptr, nullptr);
    // fea4 = relu(f4(combined)) -> fea4
    gemm_f16<0, false><<<dim3(DD2 / 128, MTOK / 128), 256, 0, stream>>>(
        ws + O_COMB, DD2, ws + O_WF4, DD2, b_f4, DD2, ws + O_FEA4, DD2, nullptr, 0, nullptr, nullptr);

    // adj = softmax(fea4 . lms^T + inf_tmp); write adj ws, adj_masked, att_loss
    adj_kernel<<<BATCH, 512, 0, stream>>>(ws + O_FEA4, ws + O_LMS, mask, adjws, out);

    // prepared = relu(f5(combined)) * lp2 -> lms (reuse)
    gemm_f16<2, false><<<dim3(DD2 / 128, MTOK / 128), 256, 0, stream>>>(
        ws + O_COMB, DD2, ws + O_WF5, DD2, b_f5, DD2, ws + O_LMS, DD2, ws + O_LP2, 0, nullptr, nullptr);

    // msg_in = adj @ prepared -> fea4 (reuse)
    msg_kernel<<<MTOK, 256, 0, stream>>>(adjws, ws + O_LMS, ws + O_FEA4);

    // messages = relu(msg_in @ W_out^T + b_out) * output_mask -> out cols 300..599
    gemm_f16<3, true><<<dim3((FEAT + 127) / 128, MTOK / 128), 256, 0, stream>>>(
        ws + O_FEA4, DD2, ws + O_WOUT, DD2, b_out, FEAT, nullptr, 0, nullptr, 0, mask, out);
}
